// StatefulConvolutionLayer_86337432584835
// MI455X (gfx1250) — matrix operand order repaired and checked by reading, not compiled, not run on hardware
//
#include <hip/hip_runtime.h>
#include <hip/hip_bf16.h>
#include <math.h>

// ---------------------------------------------------------------------------
// CDNA5 (gfx1250) wave32 WMMA types & fragment helpers
// ---------------------------------------------------------------------------
typedef __attribute__((ext_vector_type(16))) __bf16 bf16x16;
typedef __attribute__((ext_vector_type(8)))  float  floatx8;
typedef __attribute__((ext_vector_type(4)))  unsigned int u32x4;
typedef __attribute__((ext_vector_type(4)))  int    i32x4;
typedef __attribute__((ext_vector_type(8)))  int    i32x8;

#if defined(__has_builtin)
#  if __has_builtin(__builtin_amdgcn_tensor_load_to_lds) && \
      __has_builtin(__builtin_amdgcn_s_wait_tensorcnt)
#    define USE_TDM 1
#  endif
#endif
#ifndef USE_TDM
#  define USE_TDM 0
#endif

// The TDM builtin references LDS only as an integer byte-offset inside the D#
// descriptor, so the staged __shared__ array has no visible stores and its
// address never escapes -> LLVM folds the staged loads to undef and deletes
// the LDS object (rounds 2/3 proved it; a bare memory clobber was NOT enough,
// since asm can only touch memory whose address escaped to it).  Fix: leak the
// array's pointer into a volatile asm with a memory clobber, which forces the
// compiler to assume the DMA wrote through it.
__device__ __forceinline__ void publish_lds(const void* p) {
    void* q = (void*)p;
    asm volatile("" : "+v"(q) : : "memory");
}

#if USE_TDM
// Issue one Tensor-Data-Mover DMA: nelem contiguous f32 from global -> LDS.
// D# per CDNA5 ISA ch.8: group0 = {count=1, lds_addr, global_addr, type=2},
// group1 = {data_size=4B, tensor_dim0=tile_dim0=nelem, tensor_dim1=1},
// groups 2/3 zero (<=2D tensor). Wave-level op (EXEC ignored) -> call from one wave.
__device__ __forceinline__ void tdm_load_1d_f32(unsigned lds_byte_off,
                                                const void* gptr, unsigned nelem) {
    unsigned long long ga = (unsigned long long)(uintptr_t)gptr;
    u32x4 g0;
    g0[0] = 1u;                                                  // D# count = 1
    g0[1] = lds_byte_off;                                        // lds_addr
    g0[2] = (unsigned)ga;                                        // global_addr[31:0]
    g0[3] = (unsigned)((ga >> 32) & 0x01FFFFFFull) | (2u << 30); // [56:32] | type=2
    i32x8 g1;
    g1[0] = (int)(2u << 16);                       // data_size = 4 bytes
    g1[1] = (int)((nelem & 0xFFFFu) << 16);        // tensor_dim0[15:0]
    g1[2] = (int)(((nelem >> 16) & 0xFFFFu) | (1u << 16)); // dim0[31:16], tensor_dim1=1
    g1[3] = (int)((nelem & 0xFFFFu) << 16);        // tile_dim0
    g1[4] = 0;                                     // tile_dim1/2 unused
    g1[5] = (int)nelem;                            // tensor_dim0_stride
    g1[6] = 0;
    g1[7] = 0;
    i32x4 z4 = {0, 0, 0, 0};
#if __clang_major__ >= 23
    i32x8 z8 = {0, 0, 0, 0, 0, 0, 0, 0};
    __builtin_amdgcn_tensor_load_to_lds(g0, g1, z4, z4, z8, 0);
#else
    __builtin_amdgcn_tensor_load_to_lds(g0, g1, z4, z4, 0);
#endif
}
#endif

__device__ __forceinline__ floatx8 wmma_bf16(bf16x16 a, bf16x16 b, floatx8 c) {
    return __builtin_amdgcn_wmma_f32_16x16x32_bf16(
        false, a, false, b, (short)0, c, false, false);
}

// A fragment (16x32 bf16) from LDS bf16, row-major, leading dim ld.
// lanes 0-15 row M=lane, K=0..15; lanes 16-31 K=16..31 (matches B fragment)
__device__ __forceinline__ bf16x16 load_a_lds(const __bf16* base, int ld, int lane) {
    int row  = lane & 15;
    int koff = (lane >> 4) * 16;
    bf16x16 a;
#pragma unroll
    for (int e = 0; e < 16; ++e) {
        int k = koff + e;
        a[e] = base[row * ld + k];
    }
    return a;
}

// A fragment from LDS f32 (convert at fragment load)
__device__ __forceinline__ bf16x16 load_a_lds_f32(const float* base, int ld, int lane) {
    int row  = lane & 15;
    int koff = (lane >> 4) * 16;
    bf16x16 a;
#pragma unroll
    for (int e = 0; e < 16; ++e) {
        int k = koff + e;
        a[e] = (__bf16)base[row * ld + k];
    }
    return a;
}

// B fragment (32x16) from global f32 row-major (K x N) with convert.
__device__ __forceinline__ bf16x16 load_b_global_f32(const float* base, int ld, int lane) {
    int col = lane & 15;
    int k0  = (lane >> 4) * 16;
    bf16x16 b;
#pragma unroll
    for (int e = 0; e < 16; ++e)
        b[e] = (__bf16)base[(size_t)(k0 + e) * ld + col];
    return b;
}

// B fragment (32x16) from LDS bf16 row-major (K x N)
__device__ __forceinline__ bf16x16 load_b_lds(const __bf16* base, int ld, int lane) {
    int col = lane & 15;
    int k0  = (lane >> 4) * 16;
    bf16x16 b;
#pragma unroll
    for (int e = 0; e < 16; ++e)
        b[e] = base[(k0 + e) * ld + col];
    return b;
}

// D (16x16 f32): lane -> col N=lane&15; elem r -> row M = r + (lane>>4)*8
__device__ __forceinline__ void store_d_global_f32(float* base, int ld, int lane, floatx8 d) {
    int col = lane & 15;
    int m0  = (lane >> 4) * 8;
#pragma unroll
    for (int r = 0; r < 8; ++r)
        base[(size_t)(m0 + r) * ld + col] = d[r];
}

__device__ __forceinline__ void store_d_lds_bf16(__bf16* base, int ld, int lane, floatx8 d) {
    int col = lane & 15;
    int m0  = (lane >> 4) * 8;
#pragma unroll
    for (int r = 0; r < 8; ++r)
        base[(m0 + r) * ld + col] = (__bf16)d[r];
}

__device__ __forceinline__ void store_d_lds_f32(float* base, int ld, int lane, floatx8 d) {
    int col = lane & 15;
    int m0  = (lane >> 4) * 8;
#pragma unroll
    for (int r = 0; r < 8; ++r)
        base[(m0 + r) * ld + col] = d[r];
}

__device__ __forceinline__ float gelu_exact(float x) {
    return 0.5f * x * (1.0f + erff(x * 0.70710678118654752f));
}

// ---------------------------------------------------------------------------
// Kernel A: C[2048 x N] = Xw(2048x1024) @ B(1024xN), bf16 WMMA, f32 out (ws)
// block = 256 thr (8 waves): 16 rows x 128 cols; grid = (128, N/128)
// A tile staged via TDM (64KB contiguous f32).
// ---------------------------------------------------------------------------
__global__ void __launch_bounds__(256)
proj_gemm(const float* __restrict__ X, const float* __restrict__ Bw,
          float* __restrict__ C, int N) {
    __shared__ __align__(16) float sA[16 * 1024];
    const int tid = threadIdx.x, wave = tid >> 5, lane = tid & 31;
    const int row0 = blockIdx.x * 16;
    const int col0 = blockIdx.y * 128 + wave * 16;
#if USE_TDM
    if (wave == 0) {
        tdm_load_1d_f32(0u, X + (size_t)row0 * 1024, 16 * 1024);
        __builtin_amdgcn_s_wait_tensorcnt(0);
    }
    publish_lds(sA);
#else
    for (int i = tid; i < 16 * 1024; i += 256)
        sA[i] = X[(size_t)row0 * 1024 + i];
#endif
    __syncthreads();
    floatx8 acc = {};
    for (int k = 0; k < 1024; k += 32) {
        __builtin_prefetch(Bw + (size_t)(k + 32) * N + col0, 0, 1);
        bf16x16 a = load_a_lds_f32(sA + k, 1024, lane);
        bf16x16 b = load_b_global_f32(Bw + (size_t)k * N + col0, N, lane);
        acc = wmma_bf16(a, b, acc);
    }
    store_d_global_f32(C + (size_t)row0 * N + col0, N, lane, acc);
}

// ---------------------------------------------------------------------------
// Kernel B: fused  W_l = Xw@Ww_l (kept in LDS) -> M = T@W_l -> softmax(p)
//           -> g = gelu(A @ v).   block = (16 windows, one l), grid (128, 8)
// LDS (160KB total, 2 blocks co-resident per 320KB WGP):
//   [0,128K)     W_l tile, 16x4096 bf16
//   [128K,160K)  phase 2: Xw tile 16x1024 bf16
//   [128K,144K)  phase 3: T_w f32 (TDM-staged); first 512B aliased by softmax
//                stats (written only after last T_w read of the window)
//   [144K,160K)  phase 3: M f32 64x64
// ---------------------------------------------------------------------------
__global__ void __launch_bounds__(256)
window_attn(const float* __restrict__ X,  const float* __restrict__ Ww,
            const float* __restrict__ T,  const float* __restrict__ V,
            float* __restrict__ G) {
    __shared__ __align__(16) unsigned char smem[160 * 1024];
    __bf16* sW = (__bf16*)smem;                            // 16 x 4096 bf16
    __bf16* sX = (__bf16*)(smem + 128 * 1024);             // 16 x 1024 bf16
    float*  sT = (float*)(smem + 128 * 1024);              // 64 x 64 f32
    float*  sS = (float*)(smem + 128 * 1024);              // 64 max + 64 sum (alias)
    float*  sM = (float*)(smem + 144 * 1024);              // 64 x 64 f32

    const int tid = threadIdx.x, wave = tid >> 5, lane = tid & 31;
    const int w0 = blockIdx.x * 16;
    const int l  = blockIdx.y;

    // phase 1: stage Xw tile (f32 -> bf16)
    for (int i = tid; i < 16 * 1024; i += 256)
        sX[i] = (__bf16)X[(size_t)w0 * 1024 + i];
    __syncthreads();

    // phase 2: W_l tile = Xw @ Ww[:, l*4096 : (l+1)*4096] -> LDS bf16
    for (int nt = 0; nt < 4096 / (8 * 16); ++nt) {
        const int col = wave * 512 + nt * 16;
        const float* bsrc = Ww + (size_t)l * 4096 + col;
        floatx8 acc = {};
        for (int k = 0; k < 1024; k += 32) {
            __builtin_prefetch(bsrc + (size_t)(k + 32) * 32768, 0, 1);
            bf16x16 a = load_a_lds(sX + k, 1024, lane);
            bf16x16 b = load_b_global_f32(bsrc + (size_t)k * 32768, 32768, lane);
            acc = wmma_bf16(a, b, acc);
        }
        store_d_lds_bf16(sW + col, 4096, lane, acc);
    }
    __syncthreads();

    // phase 3: per window: M = T_w @ W_{l,w}, column softmax, g = gelu(A v)
    for (int w = 0; w < 16; ++w) {
        const float* Tw = T + (size_t)(w0 + w) * 4096;
#if USE_TDM
        if (wave == 0) {
            tdm_load_1d_f32(128u * 1024u, Tw, 4096);  // 16KB DMA into sT
            __builtin_amdgcn_s_wait_tensorcnt(0);
        }
        publish_lds(sT);
#else
        for (int i = tid; i < 4096; i += 256) sT[i] = Tw[i];
#endif
        __syncthreads();

        // M = T_w @ W_{l,w}: 4x4 grid of 16x16 tiles; 8 waves * 2 tiles
        for (int t2 = 0; t2 < 2; ++t2) {
            const int tile = wave * 2 + t2;
            const int mi = (tile >> 2) * 16, ni = (tile & 3) * 16;
            floatx8 acc = {};
#pragma unroll
            for (int k = 0; k < 64; k += 32) {
                bf16x16 a = load_a_lds_f32(sT + mi * 64 + k, 64, lane);
                bf16x16 b = load_b_lds(sW + w * 4096 + k * 64 + ni, 64, lane);
                acc = wmma_bf16(a, b, acc);
            }
            store_d_lds_f32(sM + mi * 64 + ni, 64, lane, acc);
        }
        __syncthreads();

        // softmax stats along p (rows) for each column r (sT dead from here on)
        if (tid < 64) {
            float mx = -3.4e38f;
            for (int p = 0; p < 64; ++p) mx = fmaxf(mx, sM[p * 64 + tid]);
            float s = 0.f;
            for (int p = 0; p < 64; ++p) s += __expf(sM[p * 64 + tid] - mx);
            sS[tid]      = mx;
            sS[64 + tid] = s;
        }
        __syncthreads();

        if (tid < 64) {
            const int p = tid;
            const float* vrow = V + (size_t)(w0 + w) * 512 + l * 64;
            float acc = 0.f;
            for (int r = 0; r < 64; ++r) {
                float a = __expf(sM[p * 64 + r] - sS[r]) / sS[64 + r];
                acc += a * vrow[r];
            }
            G[((size_t)(w0 + w) * 8 + l) * 64 + p] = gelu_exact(acc);
        }
        __syncthreads();
    }
}

// ---------------------------------------------------------------------------
// Kernel C: per-window prefix accumulation + LayerNorm.
// block = 256 thr = 8 wave32, wave l handles token (w, l); grid = 2048
// ---------------------------------------------------------------------------
__global__ void __launch_bounds__(256)
cumsum_ln(const float* __restrict__ G, const float* __restrict__ gamma,
          const float* __restrict__ beta, float* __restrict__ Y) {
    const int w = blockIdx.x;
    const int l = threadIdx.x >> 5, lane = threadIdx.x & 31;
    const float* gw = G + (size_t)w * 512;
    float y[2];
#pragma unroll
    for (int h = 0; h < 2; ++h) {
        const int p = lane + 32 * h;
        float acc = 0.f;
        if (l == 0) acc = gw[p];
        else
            for (int j = 1; j <= l; ++j) acc += gw[j * 64 + p];
        y[h] = acc;
    }
    float s = y[0] + y[1];
#pragma unroll
    for (int m = 16; m >= 1; m >>= 1) s += __shfl_xor(s, m, 32);
    const float mu = s * (1.0f / 64.0f);
    const float d0 = y[0] - mu, d1 = y[1] - mu;
    float vs = d0 * d0 + d1 * d1;
#pragma unroll
    for (int m = 16; m >= 1; m >>= 1) vs += __shfl_xor(vs, m, 32);
    const float inv = rsqrtf(vs * (1.0f / 64.0f) + 1e-5f);
    const size_t tok = (size_t)w * 8 + l;
    Y[tok * 64 + lane]      = d0 * inv * gamma[lane]      + beta[lane];
    Y[tok * 64 + lane + 32] = d1 * inv * gamma[lane + 32] + beta[lane + 32];
}

// ---------------------------------------------------------------------------
// Kernel D: FFN  out = gelu(Y@W1 + b1)@W2 + b2.  block = 16 tokens, grid 1024
// ---------------------------------------------------------------------------
__global__ void __launch_bounds__(256)
ffn(const float* __restrict__ Y, const float* __restrict__ W1,
    const float* __restrict__ B1, const float* __restrict__ W2,
    const float* __restrict__ B2, float* __restrict__ Out) {
    __shared__ __bf16 sY[16 * 64];
    __shared__ __bf16 sH[16 * 256];
    const int tid = threadIdx.x, wave = tid >> 5, lane = tid & 31;
    const int t0 = blockIdx.x * 16;
    for (int i = tid; i < 16 * 64; i += 256)
        sY[i] = (__bf16)Y[(size_t)t0 * 64 + i];
    __syncthreads();

#pragma unroll
    for (int t2 = 0; t2 < 2; ++t2) {
        const int n0 = (wave * 2 + t2) * 16;
        floatx8 acc = {};
#pragma unroll
        for (int k = 0; k < 64; k += 32) {
            bf16x16 a = load_a_lds(sY + k, 64, lane);
            bf16x16 b = load_b_global_f32(W1 + (size_t)k * 256 + n0, 256, lane);
            acc = wmma_bf16(a, b, acc);
        }
        const int col = n0 + (lane & 15);
        const float bias = B1[col];
        const int m0 = (lane >> 4) * 8;
#pragma unroll
        for (int r = 0; r < 8; ++r)
            sH[(m0 + r) * 256 + col] = (__bf16)gelu_exact(acc[r] + bias);
    }
    __syncthreads();

    if (wave < 4) {
        const int n0 = wave * 16;
        floatx8 acc = {};
#pragma unroll
        for (int k = 0; k < 256; k += 32) {
            bf16x16 a = load_a_lds(sH + k, 256, lane);
            bf16x16 b = load_b_global_f32(W2 + (size_t)k * 64 + n0, 64, lane);
            acc = wmma_bf16(a, b, acc);
        }
        const int col = n0 + (lane & 15);
        const float bias = B2[col];
        const int m0 = (lane >> 4) * 8;
#pragma unroll
        for (int r = 0; r < 8; ++r)
            Out[(size_t)(t0 + m0 + r) * 64 + col] = acc[r] + bias;
    }
}

// ---------------------------------------------------------------------------
extern "C" void kernel_launch(void* const* d_in, const int* in_sizes, int n_in,
                              void* d_out, int out_size, void* d_ws, size_t ws_size,
                              hipStream_t stream) {
    const float* x    = (const float*)d_in[0];  // (8,2048,128) == (2048,1024)
    const float* Wt   = (const float*)d_in[1];  // (1024, 4096)
    const float* Ww   = (const float*)d_in[2];  // (1024, 32768)
    const float* Wv   = (const float*)d_in[3];  // (1024, 512)
    const float* ln_g = (const float*)d_in[4];
    const float* ln_b = (const float*)d_in[5];
    const float* w1   = (const float*)d_in[6];  // (64, 256)
    const float* b1   = (const float*)d_in[7];
    const float* w2   = (const float*)d_in[8];  // (256, 64)
    const float* b2   = (const float*)d_in[9];
    float* out = (float*)d_out;                 // (8,2048,64)

    char* ws = (char*)d_ws;
    float* T = (float*)(ws);                            // 2048*4096 f32 = 32MB
    float* V = (float*)(ws + (size_t)32 * 1024 * 1024); // 2048*512  f32 = 4MB
    float* G = (float*)(ws + (size_t)36 * 1024 * 1024); // 2048*8*64 f32 = 4MB
    float* Y = (float*)(ws + (size_t)40 * 1024 * 1024); // 16384*64  f32 = 4MB

    proj_gemm<<<dim3(128, 32), 256, 0, stream>>>(x, Wt, T, 4096);
    proj_gemm<<<dim3(128, 4),  256, 0, stream>>>(x, Wv, V, 512);
    window_attn<<<dim3(128, 8), 256, 0, stream>>>(x, Ww, T, V, G);
    cumsum_ln<<<2048, 256, 0, stream>>>(G, ln_g, ln_b, Y);
    ffn<<<1024, 256, 0, stream>>>(Y, w1, b1, w2, b2, out);
}
